// DFConv2d_50182397886946
// MI455X (gfx1250) — matrix operand order repaired and checked by reading, not compiled, not run on hardware
//
#include <hip/hip_runtime.h>
#include <hip/hip_bf16.h>
#include <math.h>

typedef __attribute__((ext_vector_type(16))) _Float16 v16h;
typedef __attribute__((ext_vector_type(8)))  _Float16 v8h;
typedef __attribute__((ext_vector_type(8)))  float    v8f;

#define B_SZ   4
#define C_IN   256
#define O_OUT  256
#define HD     64
#define WD     64
#define HW     4096
#define KK     9
#define KDIM   (KK * C_IN)      // 2304
#define KSTEPS (KDIM / 32)      // 72
#define OTILES (O_OUT / 16)     // 16
#define APACK_ELEMS (OTILES * KSTEPS * 32 * 16)   // 589824 halfs
#define APACK_BYTES (APACK_ELEMS * 2)             // 1,179,648 B
#define OM_ELEMS (B_SZ * 27 * HW)                 // 442,368 floats
#define VROW 40   // padded row stride (halfs): 80 B -> 16B-aligned subrows

// ---------------------------------------------------------------------------
// Kernel 0: pack w_conv (O,C,3,3) f32 -> f16 in WMMA A-fragment layout.
// K-dim ordering: kk = k*256 + c  (k-major so each 32-step is one tap).
// ISA 16-bit A 16x32 layout: lane<16 row M=lane holds K {0..7,16..23},
// lane>=16 row M=lane-16 holds K {8..15,24..31}.
// Apack[o_tile][step][lane][e], contiguous 32B per lane.
// ---------------------------------------------------------------------------
__global__ __launch_bounds__(256) void pack_weights(
    const float* __restrict__ wconv, _Float16* __restrict__ Apack)
{
    int idx = blockIdx.x * 256 + threadIdx.x;
    if (idx >= APACK_ELEMS) return;
    int e    = idx & 15;
    int lane = (idx >> 4) & 31;
    int step = (idx >> 9) % KSTEPS;
    int ot   = idx / (KSTEPS * 512);
    int row  = lane & 15;
    int hiL  = lane >> 4;
    int kkl  = (e < 8) ? (hiL * 8 + e) : (16 + hiL * 8 + (e - 8));
    int kk   = step * 32 + kkl;       // kk = k*256 + c
    int k    = kk >> 8;
    int c    = kk & 255;
    int o    = ot * 16 + row;
    Apack[idx] = (_Float16)wconv[(o * C_IN + c) * KK + k];
}

// ---------------------------------------------------------------------------
// Kernel 1: offset conv  om(b,ch,h,w) = b_off[ch] + sum_{c,ky,kx} ...
// ---------------------------------------------------------------------------
__global__ __launch_bounds__(256) void offset_conv(
    const float* __restrict__ x, const float* __restrict__ w_off,
    const float* __restrict__ b_off, float* __restrict__ om)
{
    int idx = blockIdx.x * 256 + threadIdx.x;   // B*27*HW outputs
    if (idx >= OM_ELEMS) return;
    int w  = idx & 63;
    int h  = (idx >> 6) & 63;
    int ch = (idx >> 12) % 27;
    int b  = idx / (27 * HW);
    float acc = b_off[ch];
    const float* xb = x + (size_t)b * C_IN * HW;
    const float* wf = w_off + (size_t)ch * C_IN * KK;
    for (int c = 0; c < C_IN; ++c) {
        const float* xc = xb + c * HW;
        const float* wc = wf + c * KK;
        #pragma unroll
        for (int ky = 0; ky < 3; ++ky) {
            int yy = h + ky - 1;
            if (yy < 0 || yy >= HD) continue;
            #pragma unroll
            for (int kx = 0; kx < 3; ++kx) {
                int xx = w + kx - 1;
                if (xx < 0 || xx >= WD) continue;
                acc = fmaf(xc[yy * WD + xx], wc[ky * 3 + kx], acc);
            }
        }
    }
    om[idx] = acc;
}

// ---------------------------------------------------------------------------
// Kernel 2: fused deformable sampling + WMMA GEMM, software-pipelined.
// Block = 256 threads (8 waves), covers one image row (b,h fixed, 64 n) and
// all 256 output channels. Wave v owns o-tiles {v, v+8}.
// Double-buffered V tile: one barrier per K-step; sampling of step s+1
// overlaps WMMA of step s.
// ---------------------------------------------------------------------------
__global__ __launch_bounds__(256) void dfconv_main(
    const float* __restrict__ x,       // (B,C,64,64)
    const float* __restrict__ om,      // (B,27,64,64)
    const _Float16* __restrict__ Apack,
    float* __restrict__ out)           // (B,O,64,64)
{
    __shared__ int      sp[KK][64][4];     // clamped corner spatial index
    __shared__ float    wg[KK][64][4];     // bilinear*mask*valid weight
    __shared__ _Float16 Vt[2][64][VROW];   // double-buffered V tile [n][32 ck]

    const int tid = threadIdx.x;
    const int nb  = blockIdx.x * 64;       // global n base
    const int b   = nb >> 12;
    const int h   = (nb >> 6) & 63;

    // ---- per-(n,k) bilinear parameter precompute ----
    for (int idx = tid; idx < KK * 64; idx += 256) {
        int k = idx / 64, w = idx & 63;
        const float* omp = om + (size_t)b * 27 * HW + h * WD + w;
        float dy = omp[(2 * k) * HW];
        float dx = omp[(2 * k + 1) * HW];
        float mm = omp[(18 + k) * HW];
        mm = 1.0f / (1.0f + __expf(-mm));
        int ky = k / 3, kx = k - ky * 3;
        float py = dy + (float)(h - 1 + ky);
        float px = dx + (float)(w - 1 + kx);
        float y0f = floorf(py), x0f = floorf(px);
        float wy = py - y0f, wx = px - x0f;
        int y0 = (int)y0f, x0 = (int)x0f;
        #pragma unroll
        for (int cy = 0; cy < 2; ++cy) {
            #pragma unroll
            for (int cx = 0; cx < 2; ++cx) {
                int yi = y0 + cy, xi = x0 + cx;
                bool valid = (yi >= 0) & (yi < HD) & (xi >= 0) & (xi < WD);
                int yc = min(max(yi, 0), HD - 1);
                int xc = min(max(xi, 0), WD - 1);
                float wc = (cy ? wy : 1.0f - wy) * (cx ? wx : 1.0f - wx) * mm;
                sp[k][w][cy * 2 + cx] = yc * WD + xc;
                wg[k][w][cy * 2 + cx] = valid ? wc : 0.0f;
            }
        }
    }

    v8f acc[2][4] = {};                    // [o-tile half][n-subtile]
    const int wave = tid >> 5;
    const int lane = tid & 31;
    const int nl   = tid & 63;             // sampling n_local (w coordinate)
    const int crow = tid >> 6;             // 0..3: owns channels crow*8..crow*8+7

    const float* xb = x + (size_t)b * C_IN * HW;

    // sample step `s` into buffer `bufi`: 8 consecutive channels per thread,
    // packed into one 16B LDS store.
    auto sample = [&](int s, int bufi) {
        int k  = s >> 3;
        int c0 = (s & 7) * 32;
        int s0 = sp[k][nl][0], s1 = sp[k][nl][1];
        int s2 = sp[k][nl][2], s3 = sp[k][nl][3];
        float w0 = wg[k][nl][0], w1 = wg[k][nl][1];
        float w2 = wg[k][nl][2], w3 = wg[k][nl][3];
        const float* xc = xb + (size_t)(c0 + crow * 8) * HW;
        v8h pack;
        #pragma unroll
        for (int j = 0; j < 8; ++j) {
            float v = xc[s0] * w0 + xc[s1] * w1 + xc[s2] * w2 + xc[s3] * w3;
            pack[j] = (_Float16)v;
            xc += HW;
        }
        *(v8h*)&Vt[bufi][nl][crow * 8] = pack;
    };

    __syncthreads();                       // params visible
    sample(0, 0);                          // prologue
    int buf = 0;

    for (int step = 0; step < KSTEPS; ++step) {
        __syncthreads();                   // Vt[buf] fully written

        // A fragments for this step (issue global loads early)
        v16h a0 = *(const v16h*)(Apack + ((size_t)(wave       * KSTEPS + step) * 32 + lane) * 16);
        v16h a1 = *(const v16h*)(Apack + ((size_t)((wave + 8) * KSTEPS + step) * 32 + lane) * 16);

        // overlap: sample next step into the other buffer
        if (step + 1 < KSTEPS) sample(step + 1, buf ^ 1);

        // B fragments + WMMA from Vt[buf]
        #pragma unroll
        for (int t = 0; t < 4; ++t) {
            // lane<16: column n=t*16+lane, K0..7,16..23 ; lane>=16: same col, K8..15,24..31
            const _Float16* bp = &Vt[buf][t * 16 + (lane & 15)][(lane >> 4) * 8];
            v8h lo = *(const v8h*)(bp);
            v8h hi = *(const v8h*)(bp + 16);
            v16h bf = __builtin_shufflevector(lo, hi,
                0, 1, 2, 3, 4, 5, 6, 7, 8, 9, 10, 11, 12, 13, 14, 15);
            acc[0][t] = __builtin_amdgcn_wmma_f32_16x16x32_f16(
                false, a0, false, bf, (short)0, acc[0][t], false, false);
            acc[1][t] = __builtin_amdgcn_wmma_f32_16x16x32_f16(
                false, a1, false, bf, (short)0, acc[1][t], false, false);
        }
        buf ^= 1;
    }

    // ---- store D: VGPR r holds M=r (lanes 0-15) / M=8+r (lanes 16-31) ----
    float* ob = out + (size_t)b * O_OUT * HW + h * WD;
    const int nn   = lane & 15;
    const int mofs = (lane >> 4) * 8;
    #pragma unroll
    for (int a = 0; a < 2; ++a) {
        int obase = (wave + a * 8) * 16 + mofs;
        #pragma unroll
        for (int t = 0; t < 4; ++t) {
            int wcol = t * 16 + nn;
            #pragma unroll
            for (int r = 0; r < 8; ++r) {
                ob[(size_t)(obase + r) * HW + wcol] = acc[a][t][r];
            }
        }
    }
}

// ---------------------------------------------------------------------------
extern "C" void kernel_launch(void* const* d_in, const int* in_sizes, int n_in,
                              void* d_out, int out_size, void* d_ws, size_t ws_size,
                              hipStream_t stream) {
    const float* x      = (const float*)d_in[0];
    const float* w_off  = (const float*)d_in[1];
    const float* b_off  = (const float*)d_in[2];
    const float* w_conv = (const float*)d_in[3];
    float* out = (float*)d_out;

    _Float16* Apack = (_Float16*)d_ws;
    float*    om    = (float*)((char*)d_ws + APACK_BYTES);

    pack_weights<<<(APACK_ELEMS + 255) / 256, 256, 0, stream>>>(w_conv, Apack);
    offset_conv<<<(OM_ELEMS + 255) / 256, 256, 0, stream>>>(x, w_off, b_off, om);
    dfconv_main<<<(B_SZ * HW) / 64, 256, 0, stream>>>(x, om, Apack, out);
}
